// Channel_Attention_1898375545689
// MI455X (gfx1250) — matrix operand order repaired and checked by reading, not compiled, not run on hardware
//
#include <hip/hip_runtime.h>

// ---- CDNA5 WMMA operand types (wave32) ----
typedef __attribute__((ext_vector_type(16))) __bf16 v16bf;
typedef __attribute__((ext_vector_type(8)))  float  v8f;
typedef __attribute__((ext_vector_type(2)))  float  v2f;
typedef __attribute__((ext_vector_type(4)))  unsigned int v4u;

#define BATCH  2
#define CDIM   192
#define C2DIM  (CDIM / 2)     // 96 bf16 channel-pairs
#define CGRP   (C2DIM / 8)    // 12 groups of 8 pairs (fragment-native layout)
#define C3DIM  (3 * CDIM)     // 576
#define HDIM   256
#define WDIM   256
#define PLANE  ((size_t)HDIM * WDIM)   // 65536
#define NHEADS 6
#define DD     32             // head dim (channels per head)
#define PP     64             // pixels per 8x8 window
#define PSZ    8

static __device__ __forceinline__ unsigned int pack2bf(float lo, float hi) {
  union { __bf16 h[2]; unsigned int u; } r;
  r.h[0] = (__bf16)lo;
  r.h[1] = (__bf16)hi;
  return r.u;   // element 0 in low 16 bits (matches v16bf element order)
}

// WMMA bf16 fragment viewed as raw dwords (each dword = one K-pair).
union Frag16 {
  v16bf h;
  v4u   q[2];
  unsigned int u[8];
};

union Pack8 {
  v4u q[2];
  unsigned int u[8];
};

// =====================================================================
// Pack fp32 activations [BATCH][C][PLANE] -> fragment-native bf16 layout
//   Xq[BATCH][C/16][PLANE][8] : dword v at (g,pix) packs channels
//   (16g + 2v, 16g + 2v + 1).
// The GEMM's B-fragment (contiguous k order) takes K-pair dwords {8*lh .. 8*lh+7} of a K-32 slab at
// one pixel -> exactly 8 consecutive dwords here => 2x global_load_b128.
// =====================================================================
__global__ __launch_bounds__(256) void pack_act_bf16g(
    const float* __restrict__ in, unsigned int* __restrict__ out, int C) {
  const int    G   = C / 16;
  const size_t tot = (size_t)BATCH * G * PLANE;
  const size_t idx = (size_t)blockIdx.x * blockDim.x + threadIdx.x;
  if (idx >= tot) return;
  const size_t pix = idx % PLANE;
  const size_t g   = (idx / PLANE) % (size_t)G;
  const size_t b   = idx / ((size_t)G * PLANE);
  const float* p   = in + ((b * C + g * 16) * PLANE) + pix;
  Pack8 o;
#pragma unroll
  for (int v = 0; v < 8; ++v)
    o.u[v] = pack2bf(p[(size_t)(2 * v) * PLANE], p[(size_t)(2 * v + 1) * PLANE]);
  v4u* dst = (v4u*)(out + idx * 8);
  dst[0] = o.q[0];
  dst[1] = o.q[1];
}

// Pack 1x1-conv weights [Cout][Cin] fp32 -> [Cout][Cin/2] bf16-pair dwords
// (pairs contiguous along Cin -> A fragment = two b128 loads).
__global__ __launch_bounds__(256) void pack_w_bf16(
    const float* __restrict__ in, unsigned int* __restrict__ out, int npairs) {
  const int idx = blockIdx.x * blockDim.x + threadIdx.x;
  if (idx >= npairs) return;
  out[idx] = pack2bf(in[2 * idx], in[2 * idx + 1]);
}

// =====================================================================
// 1x1 conv as GEMM on pre-packed bf16 data:
//   Y[b][Cout][HW](f32) = W[Cout][192](bf16) * X[b][192][HW](bf16)
// Block = 128 threads = 4 waves; each wave: 16 rows x 64 cols
// (4 v_wmma_f32_16x16x32_bf16 per K-step sharing one A fragment).
// K loop: 192/32 = 6 steps; per step: 2x b128 (A) + 8x b128 (B).
// =====================================================================
__global__ __launch_bounds__(128) void gemm1x1_bf16p(
    const unsigned int* __restrict__ Xq,  // [CGRP][PLANE][8] per batch
    const unsigned int* __restrict__ Wp,  // [Cout][C2DIM]
    float* __restrict__ Y, int Cout) {
  const int lane = threadIdx.x & 31;
  const int wav  = threadIdx.x >> 5;
  const int n16  = lane & 15;
  const int lh   = lane >> 4;          // lane half
  const int row0 = blockIdx.x * 64 + wav * 16;
  const int col0 = blockIdx.y * 64;
  const int bz   = blockIdx.z;

  const unsigned int* Xb = Xq + (size_t)bz * CGRP * PLANE * 8;
  float*              Yb = Y + (size_t)bz * (size_t)Cout * PLANE;

  v8f zero = {};
  v8f acc[4];
#pragma unroll
  for (int t = 0; t < 4; ++t) acc[t] = zero;

  for (int k2 = 0; k2 < C2DIM; k2 += 16) {  // 16 K-pairs = K step of 32
    // A fragment (16x32 bf16): pair index for dword v is
    //   v + 8*lh  -> two b128 loads.
    Frag16 a;
    const unsigned int* wrow =
        Wp + (size_t)(row0 + n16) * C2DIM + k2 + 8 * lh;
    a.q[0] = *(const v4u*)(wrow);
    a.q[1] = *(const v4u*)(wrow + 4);

#pragma unroll
    for (int t = 0; t < 4; ++t) {
      // B fragment: 8 consecutive dwords of group (k2/8 + lh) at col n.
      Frag16 b;
      const unsigned int* xcol =
          Xb + ((size_t)(k2 / 8 + lh) * PLANE + (size_t)(col0 + t * 16 + n16)) * 8;
      b.q[0] = *(const v4u*)(xcol);
      b.q[1] = *(const v4u*)(xcol + 4);
      acc[t] = __builtin_amdgcn_wmma_f32_16x16x32_bf16(
          false, a.h, false, b.h, (short)0, acc[t], false, false);
    }
  }

  // D layout: VGPR r -> M = r + 8*laneHalf, N = lane&15
#pragma unroll
  for (int t = 0; t < 4; ++t) {
#pragma unroll
    for (int r = 0; r < 8; ++r) {
      Yb[(size_t)(row0 + r + 8 * lh) * PLANE + (size_t)(col0 + t * 16 + n16)] =
          acc[t][r];
    }
  }
}

// =====================================================================
// Depthwise 3x3 conv, pad 1, groups = 576. Streaming VALU kernel.
// =====================================================================
__global__ __launch_bounds__(256) void dwconv3x3(
    const float* __restrict__ in, const float* __restrict__ wdw,
    float* __restrict__ out) {
  const size_t tot = (size_t)BATCH * C3DIM * PLANE;
  const size_t idx = (size_t)blockIdx.x * blockDim.x + threadIdx.x;
  if (idx >= tot) return;
  const int    x  = (int)(idx % WDIM);
  const int    y  = (int)((idx / WDIM) % HDIM);
  const size_t bc = idx / PLANE;
  const int    ch = (int)(bc % C3DIM);
  const float* wp = wdw + ch * 9;
  const float* ip = in + bc * PLANE;
  float s = 0.f;
#pragma unroll
  for (int ky = 0; ky < 3; ++ky) {
    const int yy = y + ky - 1;
    if (yy < 0 || yy >= HDIM) continue;
#pragma unroll
    for (int kx = 0; kx < 3; ++kx) {
      const int xx = x + kx - 1;
      if (xx < 0 || xx >= WDIM) continue;
      s += wp[ky * 3 + kx] * ip[(size_t)yy * WDIM + xx];
    }
  }
  out[idx] = s;
}

// =====================================================================
// Windowed channel attention. One wave per (batch, window, head).
// q,k,v [32x64] staged in LDS; L2-normalize q,k rows; S = qn*kn^T via
// v_wmma_f32_16x16x4_f32 (fp32 matrix path, K=64); softmax; O = S*v
// (K=32). Output written directly into the fragment-native packed bf16
// layout: a lane's 4 channel-pairs per tile are 4 consecutive dwords
// => one global_store_b128 per (ti,tj).
// =====================================================================
__global__ __launch_bounds__(32) void chan_attn(
    const float* __restrict__ qkv, const float* __restrict__ temp,
    unsigned int* __restrict__ outp) {
  const int head = blockIdx.y;
  const int Wn   = WDIM / PSZ;                 // 32
  const int nwin = (HDIM / PSZ) * Wn;          // 1024
  const int b    = blockIdx.x / nwin;
  const int win  = blockIdx.x % nwin;
  const int wy   = win / Wn;
  const int wx   = win % Wn;
  const int lane = threadIdx.x;
  const int n16  = lane & 15;
  const int lh   = lane >> 4;

  __shared__ float qs[DD][PP + 1];
  __shared__ float ks[DD][PP + 1];
  __shared__ float vs[DD][PP + 1];
  __shared__ float Ss[DD][DD + 1];
  __shared__ float qin[DD];
  __shared__ float kin[DD];

  const float* qg = qkv + ((size_t)b * C3DIM + head * DD) * PLANE;
  const float* kg = qg + (size_t)CDIM * PLANE;
  const float* vg = kg + (size_t)CDIM * PLANE;

  for (int idx = lane; idx < DD * PP; idx += 32) {
    const int    i = idx >> 6;
    const int    p = idx & 63;
    const size_t g = (size_t)i * PLANE +
                     (size_t)(wy * PSZ + (p >> 3)) * WDIM + (wx * PSZ + (p & 7));
    qs[i][p] = qg[g];
    ks[i][p] = kg[g];
    vs[i][p] = vg[g];
  }
  __syncthreads();

  {
    float sq = 0.f, sk = 0.f;
#pragma unroll 8
    for (int p = 0; p < PP; ++p) {
      const float a = qs[lane][p]; sq += a * a;
      const float c = ks[lane][p]; sk += c * c;
    }
    qin[lane] = 1.f / fmaxf(sqrtf(sq), 1e-12f);
    kin[lane] = 1.f / fmaxf(sqrtf(sk), 1e-12f);
  }
  __syncthreads();

  const float tscale = temp[head];
  v8f zero = {};

  // ---- S = qn * kn^T : 32x32 in 2x2 WMMA tiles, K = 64, step 4 ----
  v8f sacc[2][2];
#pragma unroll
  for (int ti = 0; ti < 2; ++ti)
#pragma unroll
    for (int tj = 0; tj < 2; ++tj) sacc[ti][tj] = zero;

  for (int kk = 0; kk < PP; kk += 4) {
#pragma unroll
    for (int ti = 0; ti < 2; ++ti) {
      const int row = ti * 16 + n16;
      v2f a;
      a.x = qs[row][kk + 2 * lh]     * qin[row];
      a.y = qs[row][kk + 2 * lh + 1] * qin[row];
#pragma unroll
      for (int tj = 0; tj < 2; ++tj) {
        const int colr = tj * 16 + n16;
        v2f bb;
        bb.x = ks[colr][kk + 2 * lh]     * kin[colr];
        bb.y = ks[colr][kk + 2 * lh + 1] * kin[colr];
        sacc[ti][tj] = __builtin_amdgcn_wmma_f32_16x16x4_f32(
            false, a, false, bb, (short)0, sacc[ti][tj], false, false);
      }
    }
  }

#pragma unroll
  for (int ti = 0; ti < 2; ++ti)
#pragma unroll
    for (int tj = 0; tj < 2; ++tj)
#pragma unroll
      for (int r = 0; r < 8; ++r)
        Ss[ti * 16 + r + 8 * lh][tj * 16 + n16] = sacc[ti][tj][r] * tscale;
  __syncthreads();

  {
    float mx = -3.4028235e38f;
#pragma unroll 8
    for (int j = 0; j < DD; ++j) mx = fmaxf(mx, Ss[lane][j]);
    float sum = 0.f;
#pragma unroll 8
    for (int j = 0; j < DD; ++j) {
      const float e = __expf(Ss[lane][j] - mx);
      Ss[lane][j] = e;
      sum += e;
    }
    const float inv = 1.f / sum;
#pragma unroll 8
    for (int j = 0; j < DD; ++j) Ss[lane][j] *= inv;
  }
  __syncthreads();

  // ---- O = P * v : 32x64 in 2x4 WMMA tiles, K = 32, step 4 ----
  v8f oacc[2][4];
#pragma unroll
  for (int ti = 0; ti < 2; ++ti)
#pragma unroll
    for (int tj = 0; tj < 4; ++tj) oacc[ti][tj] = zero;

  for (int kk = 0; kk < DD; kk += 4) {
#pragma unroll
    for (int tj = 0; tj < 4; ++tj) {
      v2f bb;
      bb.x = vs[kk + 2 * lh][tj * 16 + n16];
      bb.y = vs[kk + 2 * lh + 1][tj * 16 + n16];
#pragma unroll
      for (int ti = 0; ti < 2; ++ti) {
        const int row = ti * 16 + n16;
        v2f a;
        a.x = Ss[row][kk + 2 * lh];
        a.y = Ss[row][kk + 2 * lh + 1];
        oacc[ti][tj] = __builtin_amdgcn_wmma_f32_16x16x4_f32(
            false, a, false, bb, (short)0, oacc[ti][tj], false, false);
      }
    }
  }

  // Store into fragment-native layout Xq[CGRP][PLANE][8]:
  // channel i = head*32 + ti*16 + 8*lh + r ; pair index = i/2 =
  // head*16 + ti*8 + 4*lh + r/2 -> group g = pidx>>3, slot v = 4*lh + r/2
  // (4 consecutive slots per lane) => one b128 store per (ti,tj).
  unsigned int* og = outp + (size_t)b * CGRP * PLANE * 8;
#pragma unroll
  for (int ti = 0; ti < 2; ++ti) {
    const int pidx0 = head * 16 + ti * 8;   // multiple of 8
    const int g     = pidx0 >> 3;
#pragma unroll
    for (int tj = 0; tj < 4; ++tj) {
      union { v4u q; unsigned int u[4]; } o;
#pragma unroll
      for (int rr = 0; rr < 4; ++rr)
        o.u[rr] = pack2bf(oacc[ti][tj][2 * rr], oacc[ti][tj][2 * rr + 1]);
      const int    p   = tj * 16 + n16;
      const size_t pix = (size_t)(wy * PSZ + (p >> 3)) * WDIM + (wx * PSZ + (p & 7));
      *(v4u*)(og + ((size_t)g * PLANE + pix) * 8 + 4 * lh) = o.q;
    }
  }
}

// =====================================================================
// Host-side launcher
// =====================================================================
extern "C" void kernel_launch(void* const* d_in, const int* in_sizes, int n_in,
                              void* d_out, int out_size, void* d_ws,
                              size_t ws_size, hipStream_t stream) {
  (void)in_sizes; (void)n_in; (void)out_size; (void)ws_size;

  const float* x           = (const float*)d_in[0];  // [2,192,256,256]
  const float* w_qkv       = (const float*)d_in[1];  // [576,192,1,1]
  const float* w_dw        = (const float*)d_in[2];  // [576,1,3,3]
  const float* w_proj      = (const float*)d_in[3];  // [192,192,1,1]
  const float* temperature = (const float*)d_in[4];  // [6,1,1]

  // Workspace: qkv_raw (302MB f32) | qkv_dw (302MB f32) | xp/midp (50MB
  // fragment-native bf16, midp aliases xp once xp is dead) | packed weights.
  float* qkv_raw = (float*)d_ws;
  float* qkv_dw  = qkv_raw + (size_t)BATCH * C3DIM * PLANE;
  unsigned int* xp     = (unsigned int*)(qkv_dw + (size_t)BATCH * C3DIM * PLANE);
  unsigned int* midp   = xp;  // alias: xp dead after GEMM 1
  unsigned int* wqkvp  = xp + (size_t)BATCH * C2DIM * PLANE;
  unsigned int* wprojp = wqkvp + (size_t)C3DIM * C2DIM;

  // 0) one-time packing to fragment-native bf16 (no in-GEMM converts,
  //    half the GEMM activation read bytes, all-b128 fragment loads).
  {
    const size_t tot = (size_t)BATCH * CGRP * PLANE;
    pack_act_bf16g<<<(unsigned)((tot + 255) / 256), 256, 0, stream>>>(x, xp,
                                                                      CDIM);
    pack_w_bf16<<<(C3DIM * C2DIM + 255) / 256, 256, 0, stream>>>(
        w_qkv, wqkvp, C3DIM * C2DIM);
    pack_w_bf16<<<(CDIM * C2DIM + 255) / 256, 256, 0, stream>>>(
        w_proj, wprojp, CDIM * C2DIM);
  }

  // 1) qkv 1x1 conv: 576x192 GEMM over 65536 pixels, per batch.
  gemm1x1_bf16p<<<dim3(C3DIM / 64, (unsigned)(PLANE / 64), BATCH), 128, 0,
                  stream>>>(xp, wqkvp, qkv_raw, C3DIM);

  // 2) depthwise 3x3.
  {
    const size_t tot = (size_t)BATCH * C3DIM * PLANE;
    dwconv3x3<<<(unsigned)((tot + 255) / 256), 256, 0, stream>>>(qkv_raw, w_dw,
                                                                 qkv_dw);
  }

  // 3) windowed channel attention: 2*1024 windows x 6 heads, 1 wave each;
  //    writes fragment-native bf16 for the proj GEMM.
  chan_attn<<<dim3(BATCH * (HDIM / PSZ) * (WDIM / PSZ), NHEADS), 32, 0,
              stream>>>(qkv_dw, temperature, midp);

  // 4) output projection 1x1 conv: 192x192 GEMM.
  gemm1x1_bf16p<<<dim3(CDIM / 64, (unsigned)(PLANE / 64), BATCH), 128, 0,
                  stream>>>(midp, wprojp, (float*)d_out, CDIM);
}
